// CrossCityAdapter_39865886441963
// MI455X (gfx1250) — matrix operand order repaired and checked by reading, not compiled, not run on hardware
//
#include <hip/hip_runtime.h>

// ---------------------------------------------------------------------------
// CrossCityAdapter fused pipeline for MI455X (gfx1250, wave32, WMMA).
//
// Roofline: the naive reference materializes pre = (B,T,S,H) f32 = 536MB.
// Fully fused, HBM traffic is ~24MB total -> the problem is latency/VALU
// bound, so we (a) never materialize pre, (b) turn the LN-variance cross
// term and the attn@s_ad einsum into v_wmma_f32_16x16x32_f16 ops, and
// (c) keep the irreducible per-pair relu-dot in LDS-tiled VALU code.
// ---------------------------------------------------------------------------

typedef __attribute__((ext_vector_type(16))) _Float16 v16h;
typedef __attribute__((ext_vector_type(8)))  _Float16 v8h;
typedef __attribute__((ext_vector_type(8)))  float    v8f;

#define H       64
#define NROW    1024      // S == T == 1024
#define NB      2
#define LN_EPS  1e-5f

__device__ __forceinline__ float sigmoidf_(float x) { return 1.0f / (1.0f + __expf(-x)); }

// ---------------------------------------------------------------------------
// Adapter: out = relu(LN(x@W1 + b1)) @ W2 + b2.   One 64-thread block per row.
// Optionally emits a transposed f16 copy (per-batch (H, NROW)) for later WMMA.
// ---------------------------------------------------------------------------
template<int D, bool WRITE_T16>
__global__ void adapter_kernel(const float* __restrict__ X,
                               const float* __restrict__ W1, const float* __restrict__ b1,
                               const float* __restrict__ g,  const float* __restrict__ beta,
                               const float* __restrict__ W2, const float* __restrict__ b2,
                               float* __restrict__ outAd, _Float16* __restrict__ outT16)
{
    __shared__ float xs[D];
    __shared__ float ys[H];
    const int row = blockIdx.x;          // 0..2047
    const int h   = threadIdx.x;         // 0..63
    const float* x = X + (size_t)row * D;
    for (int i = h; i < D; i += H) xs[i] = x[i];
    __syncthreads();

    float y = b1[h];
    #pragma unroll 8
    for (int d = 0; d < D; ++d) y = fmaf(xs[d], W1[d * H + h], y);
    ys[h] = y;
    __syncthreads();

    float sum = 0.f, ss = 0.f;
    #pragma unroll 8
    for (int k = 0; k < H; ++k) { float v = ys[k]; sum += v; ss = fmaf(v, v, ss); }
    const float m    = sum * (1.0f / H);
    const float var  = ss * (1.0f / H) - m * m;
    const float rstd = rsqrtf(var + LN_EPS);
    const float a    = fmaxf(fmaf((y - m) * rstd, g[h], beta[h]), 0.0f);
    __syncthreads();
    ys[h] = a;
    __syncthreads();

    float z = b2[h];
    #pragma unroll 8
    for (int k = 0; k < H; ++k) z = fmaf(ys[k], W2[k * H + h], z);
    outAd[(size_t)row * H + h] = z;
    if (WRITE_T16) {
        const int b = row >> 10, r = row & 1023;
        outT16[((size_t)b * H + h) * NROW + r] = (_Float16)z;
    }
}

// ---------------------------------------------------------------------------
// Project through cW half, add bias, center per row.  Emits centered rows
// (f32 + f16) and sum-of-squares of centered values (for LN variance algebra).
// ---------------------------------------------------------------------------
__global__ void proj_center_kernel(const float* __restrict__ Ad,
                                   const float* __restrict__ Wc,     // (H,H) row-major
                                   const float* __restrict__ bias,   // nullptr => 0
                                   float* __restrict__ outC,
                                   _Float16* __restrict__ outC16,
                                   float* __restrict__ outSS)
{
    __shared__ float xs[H];
    __shared__ float ys[H];
    const int row = blockIdx.x;
    const int h   = threadIdx.x;
    xs[h] = Ad[(size_t)row * H + h];
    __syncthreads();

    float y = bias ? bias[h] : 0.0f;
    #pragma unroll 8
    for (int k = 0; k < H; ++k) y = fmaf(xs[k], Wc[k * H + h], y);
    ys[h] = y;
    __syncthreads();

    float sum = 0.f, ss = 0.f;
    #pragma unroll 8
    for (int k = 0; k < H; ++k) { float v = ys[k]; sum += v; ss = fmaf(v, v, ss); }
    const float m = sum * (1.0f / H);
    const float c = y - m;
    outC[(size_t)row * H + h]   = c;
    outC16[(size_t)row * H + h] = (_Float16)c;
    if (h == 0) outSS[row] = ss - (float)H * m * m;   // sum (y-m)^2
}

// ---------------------------------------------------------------------------
// WMMA fragment builders for v_wmma_f32_16x16x32_f16 (wave32 layouts,
// ISA 7.12.2):
//   A lane L: row M=L%16, K halves {0..7,16..23} (L<16) / {8..15,24..31}
//   B lane L: col N=L%16, K halves {0..7,16..23} (L<16) / {8..15,24..31}, same as A
//   C vgpr r: (M = r + 8*(L>=16), N = L%16)
// ---------------------------------------------------------------------------
__device__ __forceinline__ v16h pack16(v8h lo, v8h hi) {
    v16h r;
    #pragma unroll
    for (int i = 0; i < 8; ++i) { r[i] = lo[i]; r[8 + i] = hi[i]; }
    return r;
}

// ---------------------------------------------------------------------------
// Fused pairwise scores kernel.  Grid (16 s-blocks, 64 t-tiles, B), 128 thr.
// Per wave: one 16x16 (t,s) tile.  WMMA computes cross = <Ut[t], Ws[s]>,
// giving LN variance without a second pass; then a single LDS VALU pass does
// relu(LN)·simW and the sigmoid.
// ---------------------------------------------------------------------------
__global__ void scores_kernel(const float* __restrict__ Ut,     // (2048,H) centered t-proj
                              const float* __restrict__ Ws,     // (2048,H) centered s-proj(+cb)
                              const _Float16* __restrict__ Uth,
                              const _Float16* __restrict__ Wsh,
                              const float* __restrict__ ssT,
                              const float* __restrict__ ssS,
                              const float* __restrict__ g,
                              const float* __restrict__ beta,
                              const float* __restrict__ simW,
                              const float* __restrict__ simb,
                              float* __restrict__ scores)       // (B,T,S)
{
    __shared__ float ldsUt[16 * 65];     // stride 65: avoid 16-way bank conflicts
    __shared__ float ldsWs[64 * 65];
    __shared__ float ldsG[H], ldsB[H], ldsW[H];

    const int b     = blockIdx.z;
    const int tTile = blockIdx.y;        // 16 t-rows
    const int sBlk  = blockIdx.x;        // 64 s-cols per block
    const int tid   = threadIdx.x;       // 0..127 (4 waves)
    const int lane  = tid & 31, wave = tid >> 5;
    const int tRow0 = b * NROW + tTile * 16;   // global row base into Ut/ssT
    const int sRow0 = b * NROW + sBlk * 64;    // global row base into Ws/ssS

    for (int i = tid; i < 16 * H; i += 128)
        ldsUt[(i >> 6) * 65 + (i & 63)] = Ut[(size_t)(tRow0 + (i >> 6)) * H + (i & 63)];
    for (int i = tid; i < 64 * H; i += 128)
        ldsWs[(i >> 6) * 65 + (i & 63)] = Ws[(size_t)(sRow0 + (i >> 6)) * H + (i & 63)];
    if (tid < H) { ldsG[tid] = g[tid]; ldsB[tid] = beta[tid]; ldsW[tid] = simW[tid]; }
    __syncthreads();

    const int mrow  = lane & 15;
    const int hiA   = (lane < 16) ? 0 : 8;
    const int koffB = (lane < 16) ? 0 : 8;

    const _Float16* aRow = Uth + (size_t)(tRow0 + mrow) * H;
    const _Float16* bRow = Wsh + (size_t)(sRow0 + wave * 16 + mrow) * H;

    v8f c = {};
    #pragma unroll
    for (int kb = 0; kb < H / 32; ++kb) {
        const int kBase = kb * 32;
        v8h alo = *(const v8h*)(aRow + kBase + hiA);
        v8h ahi = *(const v8h*)(aRow + kBase + 16 + hiA);
        v8h blo = *(const v8h*)(bRow + kBase + koffB);
        v8h bhi = *(const v8h*)(bRow + kBase + 16 + koffB);
        c = __builtin_amdgcn_wmma_f32_16x16x32_f16(
                false, pack16(alo, ahi), false, pack16(blo, bhi),
                (short)0, c, false, false);
    }

    const float sb   = simb[0];
    const int   sLoc = wave * 16 + (lane & 15);         // 0..63 within block
    const float ssSv = ssS[sRow0 + sLoc];

    #pragma unroll
    for (int r = 0; r < 8; ++r) {
        const int   tLoc = r + hiA;                     // 0..15
        const float var  = (ssT[tRow0 + tLoc] + ssSv + 2.0f * c[r]) * (1.0f / H);
        const float rstd = rsqrtf(var + LN_EPS);
        const float* u = &ldsUt[tLoc * 65];
        const float* w = &ldsWs[sLoc * 65];
        float acc = 0.f;
        #pragma unroll 8
        for (int hh = 0; hh < H; ++hh) {
            float xv = (u[hh] + w[hh]) * rstd;
            float yv = fmaxf(fmaf(xv, ldsG[hh], ldsB[hh]), 0.0f);
            acc = fmaf(yv, ldsW[hh], acc);
        }
        const int tG = tTile * 16 + tLoc;
        const int sG = sBlk * 64 + sLoc;
        scores[((size_t)b * NROW + tG) * NROW + sG] = sigmoidf_(acc + sb);
    }
}

// ---------------------------------------------------------------------------
// Row softmax over S (writes f16 attn for WMMA) + gate = sigmoid(mean(t_ad)).
// ---------------------------------------------------------------------------
__global__ void softmax_gate_kernel(const float* __restrict__ scores,
                                    const float* __restrict__ tAd,
                                    _Float16* __restrict__ attn16,
                                    float* __restrict__ gate)
{
    __shared__ float red[256];
    const int row = blockIdx.x;                 // b*1024 + t
    const int tid = threadIdx.x;                // 0..255
    const float* sr = scores + (size_t)row * NROW;

    float v[4], mx = -1e30f;
    #pragma unroll
    for (int i = 0; i < 4; ++i) { v[i] = sr[tid + i * 256]; mx = fmaxf(mx, v[i]); }
    red[tid] = mx; __syncthreads();
    for (int s = 128; s > 0; s >>= 1) { if (tid < s) red[tid] = fmaxf(red[tid], red[tid + s]); __syncthreads(); }
    mx = red[0]; __syncthreads();

    float e[4], sum = 0.f;
    #pragma unroll
    for (int i = 0; i < 4; ++i) { e[i] = __expf(v[i] - mx); sum += e[i]; }
    red[tid] = sum; __syncthreads();
    for (int s = 128; s > 0; s >>= 1) { if (tid < s) red[tid] += red[tid + s]; __syncthreads(); }
    const float inv = 1.0f / red[0];

    _Float16* ar = attn16 + (size_t)row * NROW;
    #pragma unroll
    for (int i = 0; i < 4; ++i) ar[tid + i * 256] = (_Float16)(e[i] * inv);

    if (tid == 0) {
        float s2 = 0.f;
        for (int k = 0; k < H; ++k) s2 += tAd[(size_t)row * H + k];
        gate[row] = sigmoidf_(s2 * (1.0f / H));
    }
}

// ---------------------------------------------------------------------------
// transferred = attn @ s_ad via WMMA (K=1024), fused with the gate blend.
// One 16x16 output tile per wave; 512 tiles total.
// ---------------------------------------------------------------------------
__global__ void transfer_kernel(const _Float16* __restrict__ attn16,  // (B,T,S)
                                const _Float16* __restrict__ sAdT16,  // (B,H,S)
                                const float* __restrict__ tAd,
                                const float* __restrict__ gate,
                                float* __restrict__ adapted)          // (B,T,H)
{
    const int tid  = threadIdx.x, lane = tid & 31, wave = tid >> 5;
    const int tile = blockIdx.x * 4 + wave;            // 0..511
    const int b     = tile >> 8;
    const int mTile = (tile & 255) >> 2;               // 0..63
    const int nTile = tile & 3;                        // 0..3

    const int mrow  = lane & 15;
    const int hiA   = (lane < 16) ? 0 : 8;
    const int koffB = (lane < 16) ? 0 : 8;

    const _Float16* aRow = attn16 + ((size_t)b * NROW + mTile * 16 + mrow) * NROW;
    const _Float16* bRow = sAdT16 + ((size_t)b * H + nTile * 16 + mrow) * NROW;

    v8f c = {};
    for (int kb = 0; kb < NROW / 32; ++kb) {
        const int kBase = kb * 32;
        v8h alo = *(const v8h*)(aRow + kBase + hiA);
        v8h ahi = *(const v8h*)(aRow + kBase + 16 + hiA);
        v8h blo = *(const v8h*)(bRow + kBase + koffB);
        v8h bhi = *(const v8h*)(bRow + kBase + 16 + koffB);
        c = __builtin_amdgcn_wmma_f32_16x16x32_f16(
                false, pack16(alo, ahi), false, pack16(blo, bhi),
                (short)0, c, false, false);
    }

    const int col = nTile * 16 + (lane & 15);
    #pragma unroll
    for (int r = 0; r < 8; ++r) {
        const int   t  = mTile * 16 + r + hiA;
        const float tv = tAd[((size_t)b * NROW + t) * H + col];
        const float gv = gate[b * NROW + t];
        adapted[((size_t)b * NROW + t) * H + col] = tv * (1.0f - gv) + c[r] * gv;
    }
}

// ---------------------------------------------------------------------------
extern "C" void kernel_launch(void* const* d_in, const int* in_sizes, int n_in,
                              void* d_out, int out_size, void* d_ws, size_t ws_size,
                              hipStream_t stream)
{
    const float* src   = (const float*)d_in[0];
    const float* tgt   = (const float*)d_in[1];
    const float* sW1   = (const float*)d_in[2];
    const float* sb1   = (const float*)d_in[3];
    const float* sg    = (const float*)d_in[4];
    const float* sbeta = (const float*)d_in[5];
    const float* sW2   = (const float*)d_in[6];
    const float* sb2   = (const float*)d_in[7];
    const float* tW1   = (const float*)d_in[8];
    const float* tb1   = (const float*)d_in[9];
    const float* tg    = (const float*)d_in[10];
    const float* tbeta = (const float*)d_in[11];
    const float* tW2   = (const float*)d_in[12];
    const float* tb2   = (const float*)d_in[13];
    const float* cW    = (const float*)d_in[14];
    const float* cb    = (const float*)d_in[15];
    const float* cg    = (const float*)d_in[16];
    const float* cbeta = (const float*)d_in[17];
    const float* simW  = (const float*)d_in[18];
    const float* simb  = (const float*)d_in[19];

    float* out     = (float*)d_out;
    float* adapted = out;                          // (B,T,H)  = 131072
    float* scores  = out + (size_t)NB * NROW * H;  // (B,T,S)  = 2097152

    // workspace layout (f32 region then f16 region, all 16B aligned)
    float* ws   = (float*)d_ws;
    float* s_ad = ws;                  // 131072
    float* t_ad = s_ad + 131072;       // 131072
    float* Ut   = t_ad + 131072;       // 131072
    float* Ws   = Ut   + 131072;       // 131072
    float* ssT  = Ws   + 131072;       // 2048
    float* ssS  = ssT  + 2048;         // 2048
    float* gate = ssS  + 2048;         // 2048
    _Float16* sAdT16 = (_Float16*)(gate + 2048);   // (B,H,S)   131072 halves
    _Float16* Uth    = sAdT16 + 131072;            // 131072 halves
    _Float16* Wsh    = Uth    + 131072;            // 131072 halves
    _Float16* attn16 = Wsh    + 131072;            // (B,T,S)   2097152 halves

    const int R = NB * NROW;  // 2048 rows

    adapter_kernel<128, true ><<<R, H, 0, stream>>>(src, sW1, sb1, sg, sbeta, sW2, sb2, s_ad, sAdT16);
    adapter_kernel< 64, false><<<R, H, 0, stream>>>(tgt, tW1, tb1, tg, tbeta, tW2, tb2, t_ad, nullptr);

    proj_center_kernel<<<R, H, 0, stream>>>(t_ad, cW,         nullptr, Ut, Uth, ssT);
    proj_center_kernel<<<R, H, 0, stream>>>(s_ad, cW + H * H, cb,      Ws, Wsh, ssS);

    scores_kernel<<<dim3(16, 64, NB), 128, 0, stream>>>(Ut, Ws, Uth, Wsh, ssT, ssS,
                                                        cg, cbeta, simW, simb, scores);

    softmax_gate_kernel<<<R, 256, 0, stream>>>(scores, t_ad, attn16, gate);

    transfer_kernel<<<128, 128, 0, stream>>>(attn16, sAdT16, t_ad, gate, adapted);
}
